// MultiHeadAttentionV0_15015205667429
// MI455X (gfx1250) — matrix operand order repaired and checked by reading, not compiled, not run on hardware
//
#include <hip/hip_runtime.h>
#include <hip/hip_bf16.h>
#include <stdint.h>

// Problem constants (match reference)
#define B_  4
#define S_  2048
#define D_  1024
#define H_  16
#define HD_ 64
#define NEGV (-1.0e6f)

typedef __bf16 bf16_t;
typedef bf16_t v16bf __attribute__((ext_vector_type(16)));
typedef bf16_t v2bf  __attribute__((ext_vector_type(2)));
typedef float  v8f   __attribute__((ext_vector_type(8)));

union Frag {
    v16bf    v;
    uint32_t u[8];
};

// Native f32 -> bf16 (RNE) via hardware cvt
__device__ __forceinline__ uint16_t f2bf(float f) {
    bf16_t   h = (bf16_t)f;
    uint16_t r;
    __builtin_memcpy(&r, &h, 2);
    return r;
}
__device__ __forceinline__ uint32_t pack2(float lo, float hi) {
    v2bf p;
    p.x = (bf16_t)lo;
    p.y = (bf16_t)hi;
    uint32_t r;
    __builtin_memcpy(&r, &p, 4);
    return r;
}
// 16-bit A-matrix 16x32 k-pattern (ISA 7.12.2): VGPR v holds K pair (k,k+1)
__device__ __forceinline__ int kpatA(int v, int hl) {
    return 2 * v + (v >= 4 ? 8 : 0) + hl * 8;
}
// 16-bit B-matrix 32x16 k-pattern: same K order as A (VGPR v holds K pair (k,k+1))
__device__ __forceinline__ int kpatB(int v, int hl) {
    return 2 * v + (v >= 4 ? 8 : 0) + hl * 8;
}
__device__ __forceinline__ v8f wmma_bf16(const Frag& a, const Frag& b, v8f c) {
    return __builtin_amdgcn_wmma_f32_16x16x32_bf16(false, a.v, false, b.v,
                                                   (short)0, c, false, false);
}
// CDNA5 async copy: 16 bytes global -> LDS, tracked on ASYNCcnt
__device__ __forceinline__ void async_copy16(uint32_t lds_off, const void* g) {
    asm volatile("global_load_async_to_lds_b128 %0, %1, off"
                 :
                 : "v"(lds_off), "v"((uint64_t)(uintptr_t)g)
                 : "memory");
}
__device__ __forceinline__ void wait_async0() {
    asm volatile("s_wait_asynccnt 0x0" ::: "memory");
}
__device__ __forceinline__ uint32_t lds_off_of(const void* p) {
    return (uint32_t)(uintptr_t)p;   // generic LDS pointer: low 32 bits = LDS addr
}

// ---------------------------------------------------------------------------
// GEMM: C[M,N] = A[M,K] * W[K,N]   (M=8192, K=N=1024)
// A_BF16: A is bf16 (pure copy -> async-to-LDS); else f32 converted in staging
// OUT_SPLIT: write bf16 head-split [B,H,S,64]; else f32 row-major [M,N]
// ---------------------------------------------------------------------------
template <bool A_BF16, bool OUT_SPLIT>
__global__ __launch_bounds__(256) void gemm_wmma(const void* __restrict__ Aptr,
                                                 const float* __restrict__ W,
                                                 void* __restrict__ Cout,
                                                 float scale) {
    constexpr int Kd = D_;
    constexpr int N  = D_;
    __shared__ __align__(16) uint16_t As[128 * 40];   // [m][k] stride 40
    __shared__ __align__(16) uint16_t Bs[128 * 40];   // [n][k] (transposed) stride 40

    const int tid  = threadIdx.x;
    const int lane = tid & 31;
    const int wave = tid >> 5;
    const int hl   = lane >> 4;       // half of wave
    const int l16  = lane & 15;

    const int bm0 = blockIdx.y * 128;
    const int bn0 = blockIdx.x * 128;
    const int wm  = (wave & 3) * 32;  // wave row offset within block tile
    const int wn  = (wave >> 2) * 64; // wave col offset within block tile

    v8f acc[2][4] = {};

    const int arow_l = tid >> 1;           // 0..127
    const int aks    = (tid & 1) * 16;     // 0 or 16
    const int bk     = tid >> 3;           // 0..31
    const int bn     = (tid & 7) * 16;     // 0..112

    for (int kb = 0; kb < Kd; kb += 32) {
        // ---- stage A tile (128 x 32) as bf16 ----
        if (A_BF16) {
            const uint16_t* A16 = (const uint16_t*)Aptr;
            const uint16_t* src = A16 + (size_t)(bm0 + arow_l) * Kd + kb + aks;
            const uint32_t  dst = lds_off_of(&As[arow_l * 40 + aks]);
            async_copy16(dst, src);
            async_copy16(dst + 16, src + 8);
        } else {
            const float* A32 = (const float*)Aptr;
            const float4* src =
                (const float4*)(A32 + (size_t)(bm0 + arow_l) * Kd + kb + aks);
            uint32_t* dst = (uint32_t*)&As[arow_l * 40 + aks];
#pragma unroll
            for (int i = 0; i < 4; i++) {
                float4 f       = src[i];
                dst[i * 2 + 0] = pack2(f.x, f.y);
                dst[i * 2 + 1] = pack2(f.z, f.w);
            }
        }
        // ---- stage W tile (32 x 128) transposed -> Bs[n][k] ----
        {
            const float4* src =
                (const float4*)(W + (size_t)(kb + bk) * N + bn0 + bn);
#pragma unroll
            for (int i = 0; i < 4; i++) {
                float4 f                       = src[i];
                Bs[(bn + i * 4 + 0) * 40 + bk] = f2bf(f.x);
                Bs[(bn + i * 4 + 1) * 40 + bk] = f2bf(f.y);
                Bs[(bn + i * 4 + 2) * 40 + bk] = f2bf(f.z);
                Bs[(bn + i * 4 + 3) * 40 + bk] = f2bf(f.w);
            }
        }
        // prefetch next k-tiles while we compute on this one
        if (kb + 32 < Kd) {
            if (!A_BF16)
                __builtin_prefetch((const float*)Aptr +
                                       (size_t)(bm0 + arow_l) * Kd + kb + 32 + aks,
                                   0, 1);
            __builtin_prefetch(W + (size_t)(kb + 32 + bk) * N + bn0 + bn, 0, 1);
        }
        if (A_BF16) wait_async0();
        __syncthreads();

        Frag af[2], bfr[4];
#pragma unroll
        for (int i = 0; i < 2; i++) {
            const int row = wm + i * 16 + l16;
#pragma unroll
            for (int v = 0; v < 8; v++)
                af[i].u[v] = *(const uint32_t*)&As[row * 40 + kpatA(v, hl)];
        }
#pragma unroll
        for (int j = 0; j < 4; j++) {
            const int col = wn + j * 16 + l16;
#pragma unroll
            for (int v = 0; v < 8; v++)
                bfr[j].u[v] = *(const uint32_t*)&Bs[col * 40 + kpatB(v, hl)];
        }
#pragma unroll
        for (int i = 0; i < 2; i++)
#pragma unroll
            for (int j = 0; j < 4; j++)
                acc[i][j] = wmma_bf16(af[i], bfr[j], acc[i][j]);
        __syncthreads();
    }

    // ---- write C ----
#pragma unroll
    for (int i = 0; i < 2; i++) {
#pragma unroll
        for (int j = 0; j < 4; j++) {
#pragma unroll
            for (int r = 0; r < 8; r++) {
                const int   m   = bm0 + wm + i * 16 + hl * 8 + r;
                const int   n   = bn0 + wn + j * 16 + l16;
                const float val = acc[i][j][r] * scale;
                if (OUT_SPLIT) {
                    const int b = m >> 11;        // /S_
                    const int q = m & (S_ - 1);
                    const int h = n >> 6;         // /HD_
                    const int d = n & (HD_ - 1);
                    ((uint16_t*)Cout)[(((size_t)(b * H_ + h)) * S_ + q) * HD_ + d] =
                        f2bf(val);
                } else {
                    ((float*)Cout)[(size_t)m * N + n] = val;
                }
            }
        }
    }
}

// ---------------------------------------------------------------------------
// Flash attention over head-split bf16 Q/K/V. Q pre-scaled by 1/sqrt(hd).
// Grid: (S/128, B*H). Block: 256 threads (8 waves, 16 q-rows each).
// ---------------------------------------------------------------------------
__global__ __launch_bounds__(256) void flash_attn(const uint16_t* __restrict__ Qh,
                                                  const uint16_t* __restrict__ Kh,
                                                  const uint16_t* __restrict__ Vh,
                                                  const int* __restrict__ vlens,
                                                  uint16_t* __restrict__ Om) {
    __shared__ __align__(16) uint16_t Ks[32 * 72];       // [key][d] stride 72
    __shared__ __align__(16) uint16_t Vt[64 * 40];       // [d][key] stride 40
    __shared__ __align__(16) uint16_t Ps[8][16 * 40];    // per-wave P tile

    const int tid  = threadIdx.x;
    const int lane = tid & 31;
    const int wave = tid >> 5;
    const int hl   = lane >> 4;
    const int l16  = lane & 15;

    const int    bh    = blockIdx.y;          // b*H + h
    const int    b     = bh >> 4;
    const int    h     = bh & 15;
    const size_t offs  = (size_t)bh * S_ * HD_;
    const int    qrow0 = blockIdx.x * 128 + wave * 16;
    const int    vlen  = vlens[b];

    // ---- load Q fragments once (rows qrow0..+15, k = 0..63) ----
    Frag qf[2];
#pragma unroll
    for (int c = 0; c < 2; c++)
#pragma unroll
        for (int v = 0; v < 8; v++) {
            const int row = qrow0 + l16;
            const int k   = kpatA(v, hl) + c * 32;
            qf[c].u[v] = *(const uint32_t*)(Qh + offs + (size_t)row * HD_ + k);
        }

    float mrow[8], lrow[8];
    v8f   o[4] = {};
#pragma unroll
    for (int r = 0; r < 8; r++) {
        mrow[r] = -1e30f;
        lrow[r] = 0.0f;
    }

    const int skey = tid >> 3;        // 0..31 (staging key row)
    const int sd0  = (tid & 7) * 8;   // 0..56 (staging d segment)

    for (int kb = 0; kb < S_; kb += 32) {
        if (kb >= vlen) break;   // fully masked block: exp underflows to exactly 0
        __syncthreads();
        // ---- stage K block [32 x 64] via CDNA5 async global->LDS copy ----
        async_copy16(lds_off_of(&Ks[skey * 72 + sd0]),
                     Kh + offs + (size_t)(kb + skey) * HD_ + sd0);
        // ---- stage V block transposed -> Vt[d][key] (needs lane transpose) ----
        {
            const uint4* src =
                (const uint4*)(Vh + offs + (size_t)(kb + skey) * HD_ + sd0);
            uint4           p  = src[0];
            const uint16_t* ps = (const uint16_t*)&p;
#pragma unroll
            for (int i = 0; i < 8; i++) Vt[(sd0 + i) * 40 + skey] = ps[i];
        }
        wait_async0();
        __syncthreads();

        // ---- scores S = Q * K^T (16 x 32), two 16x16 tiles ----
        v8f s[2] = {};
#pragma unroll
        for (int t = 0; t < 2; t++) {
#pragma unroll
            for (int c = 0; c < 2; c++) {
                Frag      kf;
                const int keyrow = t * 16 + l16;
#pragma unroll
                for (int v = 0; v < 8; v++)
                    kf.u[v] = *(const uint32_t*)&Ks[keyrow * 72 +
                                                    kpatB(v, hl) + c * 32];
                s[t] = wmma_bf16(qf[c], kf, s[t]);
            }
            // column mask: key index >= valid_len -> NEG fill
            const int keyg = kb + t * 16 + l16;
            if (keyg >= vlen) {
#pragma unroll
                for (int r = 0; r < 8; r++) s[t][r] = NEGV;
            }
        }

        // ---- online softmax (row = hl*8 + r, reduce across 16-lane half) ----
        float corr[8];
#pragma unroll
        for (int r = 0; r < 8; r++) {
            float rm = fmaxf(s[0][r], s[1][r]);
#pragma unroll
            for (int msk = 1; msk < 16; msk <<= 1)
                rm = fmaxf(rm, __shfl_xor(rm, msk, 32));
            const float mnew = fmaxf(mrow[r], rm);
            corr[r]          = __expf(mrow[r] - mnew);
            mrow[r]          = mnew;
            s[0][r]          = __expf(s[0][r] - mnew);
            s[1][r]          = __expf(s[1][r] - mnew);
            float rs         = s[0][r] + s[1][r];
#pragma unroll
            for (int msk = 1; msk < 16; msk <<= 1)
                rs += __shfl_xor(rs, msk, 32);
            lrow[r] = lrow[r] * corr[r] + rs;
        }
#pragma unroll
        for (int j = 0; j < 4; j++)
#pragma unroll
            for (int r = 0; r < 8; r++) o[j][r] = o[j][r] * corr[r];

        // ---- C-layout P -> bf16 A-layout via per-wave LDS tile ----
        uint16_t* ps = &Ps[wave][0];
#pragma unroll
        for (int r = 0; r < 8; r++) {
            const int mloc           = hl * 8 + r;
            ps[mloc * 40 + l16]      = f2bf(s[0][r]);
            ps[mloc * 40 + 16 + l16] = f2bf(s[1][r]);
        }
        Frag pf;
#pragma unroll
        for (int v = 0; v < 8; v++)
            pf.u[v] = *(const uint32_t*)&ps[l16 * 40 + kpatA(v, hl)];

        // ---- O += P * V (4 tiles over d) ----
#pragma unroll
        for (int j = 0; j < 4; j++) {
            Frag      vf;
            const int d = j * 16 + l16;
#pragma unroll
            for (int v = 0; v < 8; v++)
                vf.u[v] = *(const uint32_t*)&Vt[d * 40 + kpatB(v, hl)];
            o[j] = wmma_bf16(pf, vf, o[j]);
        }
    }

    // ---- finalize: O /= l, write merged-head bf16 [B,S,D] ----
#pragma unroll
    for (int r = 0; r < 8; r++) lrow[r] = 1.0f / lrow[r];
#pragma unroll
    for (int j = 0; j < 4; j++)
#pragma unroll
        for (int r = 0; r < 8; r++) {
            const int q = qrow0 + hl * 8 + r;
            const int n = h * HD_ + j * 16 + l16;
            Om[((size_t)b * S_ + q) * D_ + n] = f2bf(o[j][r] * lrow[r]);
        }
}

// ---------------------------------------------------------------------------
extern "C" void kernel_launch(void* const* d_in, const int* in_sizes, int n_in,
                              void* d_out, int out_size, void* d_ws,
                              size_t ws_size, hipStream_t stream) {
    (void)in_sizes; (void)n_in; (void)out_size; (void)ws_size;
    const float* queries = (const float*)d_in[0];
    const float* keys    = (const float*)d_in[1];
    const float* values  = (const float*)d_in[2];
    const int*   vlens   = (const int*)d_in[3];
    const float* Wq      = (const float*)d_in[4];
    const float* Wk      = (const float*)d_in[5];
    const float* Wv      = (const float*)d_in[6];
    const float* Wo      = (const float*)d_in[7];
    float*       out     = (float*)d_out;

    const size_t headElems = (size_t)B_ * H_ * S_ * HD_;   // 8M elems
    uint16_t*    Qh        = (uint16_t*)d_ws;
    uint16_t*    Kh        = Qh + headElems;
    uint16_t*    Vh        = Kh + headElems;
    uint16_t*    Om        = Vh + headElems;

    dim3 blk(256);
    dim3 gGemm(D_ / 128, (B_ * S_) / 128);   // (8, 64)
    dim3 gAttn(S_ / 128, B_ * H_);           // (16, 64)

    const float qscale = 0.125f;   // 1/sqrt(HD_)
    gemm_wmma<false, true><<<gGemm, blk, 0, stream>>>(queries, Wq, Qh, qscale);
    gemm_wmma<false, true><<<gGemm, blk, 0, stream>>>(keys,    Wk, Kh, 1.0f);
    gemm_wmma<false, true><<<gGemm, blk, 0, stream>>>(values,  Wv, Vh, 1.0f);

    flash_attn<<<gAttn, blk, 0, stream>>>(Qh, Kh, Vh, vlens, Om);

    gemm_wmma<true, false><<<gGemm, blk, 0, stream>>>(Om, Wo, out, 1.0f);
}
